// GATNet_28913719837234
// MI455X (gfx1250) — matrix operand order repaired and checked by reading, not compiled, not run on hardware
//
#include <hip/hip_runtime.h>
#include <hip/hip_bf16.h>
#include <math.h>

typedef __attribute__((ext_vector_type(16))) __bf16 v16bf;
typedef __attribute__((ext_vector_type(8)))  float  v8f;

#define N_NODES  20000
#define N_EDGES  200000
#define N_GRAPHS 32
#define STAT_CHUNK 256
#define POOL_CHUNK 256

static __device__ __forceinline__ float sigmoidf_(float x) {
  return 1.0f / (1.0f + __expf(-x));
}

// ----------------------------------------------------------------- zero
__global__ void zero_f32(float* __restrict__ p, int n) {
  int i = blockIdx.x * blockDim.x + threadIdx.x;
  if (i < n) p[i] = 0.0f;
}

// --------------------------------------------- fp32 -> bf16, K padded to 32
// grid: (N_NODES, ceil(Kpad/256))
__global__ void cvt_h_bf16(const float* __restrict__ H, __bf16* __restrict__ Hb,
                           int d, int Kpad) {
  int n = blockIdx.x;
  int k = blockIdx.y * blockDim.x + threadIdx.x;
  if (k >= Kpad) return;
  float v = (k < d) ? H[(size_t)n * d + k] : 0.0f;
  Hb[(size_t)n * Kpad + k] = (__bf16)v;
}

// ------------------- weight convert + transpose: W[dIn x dOut] -> Wt[Npad x Kpad]
// grid: (Npad, ceil(Kpad/256))
__global__ void cvt_w_bf16t(const float* __restrict__ W, __bf16* __restrict__ Wt,
                            int dIn, int dOut, int Kpad) {
  int nrow = blockIdx.x;
  int k = blockIdx.y * blockDim.x + threadIdx.x;
  if (k >= Kpad) return;
  float v = (k < dIn && nrow < dOut) ? W[(size_t)k * dOut + nrow] : 0.0f;
  Wt[(size_t)nrow * Kpad + k] = (__bf16)v;
}

// ----------------------------------------------------------------- WMMA GEMM
// C[M x N] = A[M x Kpad](bf16) * Bt[Npad x Kpad]^T(bf16) + bias[N]
// Block: 256 threads = 8 waves, arranged 2(M) x 4(N).
// Wave tile 32(M) x 64(N): 2 A frags, 4 B frags, 8 accumulators ->
// 8 v_wmma per K-step from 12 b128 loads (1.5 loads/wmma).
__global__ __launch_bounds__(256) void gemm_wmma_bf16(
    const __bf16* __restrict__ A,
    const __bf16* __restrict__ Bt,
    const float*  __restrict__ bias,
    float* __restrict__ C,
    int M, int N, int Kpad) {
  const int lane = threadIdx.x & 31;
  const int wave = threadIdx.x >> 5;
  const int wm = wave & 1;                 // 2 M-waves
  const int wn = wave >> 1;                // 4 N-waves
  const int m0 = blockIdx.y * 64 + wm * 32;
  const int n0 = blockIdx.x * 256 + wn * 64;
  if (m0 + 32 > M) return;   // wave-uniform (M % 32 == 0 here), EXEC stays full
  const int l16 = lane & 15;
  const int kh  = lane >> 4;

  // A fragment (16x32 bf16): contiguous 16 k's starting at k0 + kh*16
  const __bf16* ap0 = A + (size_t)(m0 + l16) * Kpad + kh * 16;
  const __bf16* ap1 = ap0 + (size_t)16 * Kpad;
  // B fragment (32x16 bf16): lane column n, contiguous 16 k's starting at kh*16
  const __bf16* bp0 = Bt + (size_t)(n0 +  0 + l16) * Kpad + kh * 16;
  const __bf16* bp1 = Bt + (size_t)(n0 + 16 + l16) * Kpad + kh * 16;
  const __bf16* bp2 = Bt + (size_t)(n0 + 32 + l16) * Kpad + kh * 16;
  const __bf16* bp3 = Bt + (size_t)(n0 + 48 + l16) * Kpad + kh * 16;

  v8f acc[8] = {};   // acc[t*2 + mi] : tile t (N), sub-tile mi (M)

  for (int k0 = 0; k0 < Kpad; k0 += 32) {
    union Frag { v16bf v; uint4 u[2]; } fa0, fa1, f0, f1, f2, f3;
    fa0.u[0] = *(const uint4*)(ap0 + k0);
    fa0.u[1] = *(const uint4*)(ap0 + k0 + 8);
    fa1.u[0] = *(const uint4*)(ap1 + k0);
    fa1.u[1] = *(const uint4*)(ap1 + k0 + 8);
    f0.u[0]  = *(const uint4*)(bp0 + k0);
    f0.u[1]  = *(const uint4*)(bp0 + k0 + 8);
    f1.u[0]  = *(const uint4*)(bp1 + k0);
    f1.u[1]  = *(const uint4*)(bp1 + k0 + 8);
    f2.u[0]  = *(const uint4*)(bp2 + k0);
    f2.u[1]  = *(const uint4*)(bp2 + k0 + 8);
    f3.u[0]  = *(const uint4*)(bp3 + k0);
    f3.u[1]  = *(const uint4*)(bp3 + k0 + 8);

    // prefetch the streaming rows a little ahead (global_prefetch_b8)
    if (k0 + 128 < Kpad) {
      __builtin_prefetch(ap0 + k0 + 128, 0, 1);
      __builtin_prefetch(bp0 + k0 + 128, 0, 1);
    }

    acc[0] = __builtin_amdgcn_wmma_f32_16x16x32_bf16(false, fa0.v, false, f0.v, (short)0, acc[0], false, false);
    acc[1] = __builtin_amdgcn_wmma_f32_16x16x32_bf16(false, fa1.v, false, f0.v, (short)0, acc[1], false, false);
    acc[2] = __builtin_amdgcn_wmma_f32_16x16x32_bf16(false, fa0.v, false, f1.v, (short)0, acc[2], false, false);
    acc[3] = __builtin_amdgcn_wmma_f32_16x16x32_bf16(false, fa1.v, false, f1.v, (short)0, acc[3], false, false);
    acc[4] = __builtin_amdgcn_wmma_f32_16x16x32_bf16(false, fa0.v, false, f2.v, (short)0, acc[4], false, false);
    acc[5] = __builtin_amdgcn_wmma_f32_16x16x32_bf16(false, fa1.v, false, f2.v, (short)0, acc[5], false, false);
    acc[6] = __builtin_amdgcn_wmma_f32_16x16x32_bf16(false, fa0.v, false, f3.v, (short)0, acc[6], false, false);
    acc[7] = __builtin_amdgcn_wmma_f32_16x16x32_bf16(false, fa1.v, false, f3.v, (short)0, acc[7], false, false);
  }

#pragma unroll
  for (int t = 0; t < 4; ++t) {
    int n = n0 + t * 16 + l16;
    if (n < N) {                         // tail guard for d_out = 516
      float b = bias[n];
#pragma unroll
      for (int mi = 0; mi < 2; ++mi) {
#pragma unroll
        for (int r = 0; r < 8; ++r) {
          int m = m0 + mi * 16 + kh * 8 + r;  // C/D layout: VGPR r -> row r (+8 hi half)
          C[(size_t)m * N + n] = acc[t * 2 + mi][r] + b;
        }
      }
    }
  }
}

// ----------------------------------------------------------------- edges
// agg[dst] += sigmoid(K[dst] + Q[src]) * V[src]; one block per edge
__global__ void edge_msg(const int* __restrict__ ei,
                         const float* __restrict__ Kf, const float* __restrict__ Qf,
                         const float* __restrict__ Vf, float* __restrict__ agg, int d) {
  int e = blockIdx.x;
  int src = ei[e];
  int dst = ei[N_EDGES + e];
  const float* kr = Kf + (size_t)dst * d;
  const float* qr = Qf + (size_t)src * d;
  const float* vr = Vf + (size_t)src * d;
  float* ar = agg + (size_t)dst * d;
  for (int f = threadIdx.x; f < d; f += blockDim.x) {
    float g = sigmoidf_(kr[f] + qr[f]);
    __hip_atomic_fetch_add(&ar[f], g * vr[f], __ATOMIC_RELAXED, __HIP_MEMORY_SCOPE_AGENT);
  }
}

// ----------------------------------------------------------------- batch norm
// stats of relu(X) per feature: st[0..d)=sum, st[d..2d)=sumsq
__global__ void bn_stats(const float* __restrict__ X, float* __restrict__ st, int d) {
  int f = blockIdx.x * blockDim.x + threadIdx.x;
  if (f >= d) return;
  int n0 = blockIdx.y * STAT_CHUNK;
  int n1 = min(n0 + STAT_CHUNK, N_NODES);
  float s = 0.0f, s2 = 0.0f;
  for (int n = n0; n < n1; ++n) {
    float v = fmaxf(X[(size_t)n * d + f], 0.0f);
    s += v; s2 += v * v;
  }
  __hip_atomic_fetch_add(&st[f],     s,  __ATOMIC_RELAXED, __HIP_MEMORY_SCOPE_AGENT);
  __hip_atomic_fetch_add(&st[d + f], s2, __ATOMIC_RELAXED, __HIP_MEMORY_SCOPE_AGENT);
}

__global__ void bn_finalize(float* __restrict__ st, const float* __restrict__ gamma,
                            const float* __restrict__ beta, int d) {
  int f = blockIdx.x * blockDim.x + threadIdx.x;
  if (f >= d) return;
  const float invN = 1.0f / (float)N_NODES;
  float mean = st[f] * invN;
  float var  = st[d + f] * invN - mean * mean;
  float sc   = gamma[f] * rsqrtf(var + 1e-5f);
  st[2 * d + f] = sc;
  st[3 * d + f] = beta[f] - mean * sc;
}

// relu + affine; writes fp32 H and bf16 Hb (padded for next layer's GEMM)
// grid: (N_NODES, ceil(KpadNext/256))
__global__ void bn_apply(const float* __restrict__ X, const float* __restrict__ st,
                         float* __restrict__ H, __bf16* __restrict__ Hb,
                         int d, int KpadNext) {
  int n = blockIdx.x;
  int k = blockIdx.y * blockDim.x + threadIdx.x;
  if (k >= KpadNext) return;
  float y = 0.0f;
  if (k < d) {
    float v = fmaxf(X[(size_t)n * d + k], 0.0f);
    y = v * st[2 * d + k] + st[3 * d + k];
    H[(size_t)n * d + k] = y;
  }
  Hb[(size_t)n * KpadNext + k] = (__bf16)y;
}

// ----------------------------------------------------------------- pooling
// batch is sorted: run-length accumulate in registers, flush on graph change
__global__ void pool_sum(const float* __restrict__ H, const int* __restrict__ batch,
                         float* __restrict__ out, int d) {
  int f = blockIdx.x * blockDim.x + threadIdx.x;
  if (f >= d) return;
  int n0 = blockIdx.y * POOL_CHUNK;
  int n1 = min(n0 + POOL_CHUNK, N_NODES);
  int cur = batch[n0];
  float acc = 0.0f;
  for (int n = n0; n < n1; ++n) {
    int g = batch[n];
    if (g != cur) {
      __hip_atomic_fetch_add(&out[(size_t)cur * d + f], acc, __ATOMIC_RELAXED, __HIP_MEMORY_SCOPE_AGENT);
      acc = 0.0f; cur = g;
    }
    acc += H[(size_t)n * d + f];
  }
  __hip_atomic_fetch_add(&out[(size_t)cur * d + f], acc, __ATOMIC_RELAXED, __HIP_MEMORY_SCOPE_AGENT);
}

// ----------------------------------------------------------------- driver
extern "C" void kernel_launch(void* const* d_in, const int* in_sizes, int n_in,
                              void* d_out, int out_size, void* d_ws, size_t ws_size,
                              hipStream_t stream) {
  (void)in_sizes; (void)n_in; (void)ws_size;
  const float* x     = (const float*)d_in[0];
  const int*   ei    = (const int*)d_in[1];
  const int*   batch = (const int*)d_in[2];

  const int dIn[5]  = {128, 1024, 512, 256, 516};
  const int dOut[5] = {1024, 512, 256, 516, 1024};
  auto rnd = [](int v, int m) { return (v + m - 1) / m * m; };

  // workspace carve (256B aligned)
  char* ws = (char*)d_ws;
  size_t off = 0;
  auto carve = [&](size_t bytes) -> void* {
    void* p = ws + off;
    off += (bytes + 255) & ~(size_t)255;
    return p;
  };
  __bf16* Hb = (__bf16*)carve((size_t)N_NODES * 1024 * sizeof(__bf16)); // bf16 activations (padded)
  __bf16* Wt[4];
  for (int i = 0; i < 4; ++i)
    Wt[i] = (__bf16*)carve((size_t)1024 * 1024 * sizeof(__bf16));       // transposed bf16 weights
  float* Kb = (float*)carve((size_t)N_NODES * 1024 * sizeof(float));
  float* Qb = (float*)carve((size_t)N_NODES * 1024 * sizeof(float));
  float* Vb = (float*)carve((size_t)N_NODES * 1024 * sizeof(float));
  float* Ag = (float*)carve((size_t)N_NODES * 1024 * sizeof(float));    // skip + aggregated messages
  float* H  = (float*)carve((size_t)N_NODES * 1024 * sizeof(float));    // fp32 activations
  float* st = (float*)carve((size_t)4 * 1024 * sizeof(float));          // sum/sumsq/scale/shift

  // initial x -> bf16
  {
    int d = dIn[0], Kpad = rnd(d, 32);
    cvt_h_bf16<<<dim3(N_NODES, (Kpad + 255) / 256), 256, 0, stream>>>(x, Hb, d, Kpad);
  }

  for (int L = 0; L < 5; ++L) {
    const int di = dIn[L], dn = dOut[L];
    const int Kpad = rnd(di, 32);
    const int Npad = rnd(dn, 256);
    const float* P[10];
    for (int j = 0; j < 10; ++j) P[j] = (const float*)d_in[3 + L * 10 + j];
    const float *Wk = P[0], *bk = P[1], *Wq = P[2], *bq = P[3],
                *Wv = P[4], *bv = P[5], *Ws2 = P[6], *bs = P[7],
                *gamma = P[8], *beta = P[9];

    dim3 gw(Npad, (Kpad + 255) / 256);
    cvt_w_bf16t<<<gw, 256, 0, stream>>>(Wk,  Wt[0], di, dn, Kpad);
    cvt_w_bf16t<<<gw, 256, 0, stream>>>(Wq,  Wt[1], di, dn, Kpad);
    cvt_w_bf16t<<<gw, 256, 0, stream>>>(Wv,  Wt[2], di, dn, Kpad);
    cvt_w_bf16t<<<gw, 256, 0, stream>>>(Ws2, Wt[3], di, dn, Kpad);

    dim3 gg(Npad / 256, (N_NODES + 63) / 64);
    gemm_wmma_bf16<<<gg, 256, 0, stream>>>(Hb, Wt[0], bk, Kb, N_NODES, dn, Kpad);
    gemm_wmma_bf16<<<gg, 256, 0, stream>>>(Hb, Wt[1], bq, Qb, N_NODES, dn, Kpad);
    gemm_wmma_bf16<<<gg, 256, 0, stream>>>(Hb, Wt[2], bv, Vb, N_NODES, dn, Kpad);
    gemm_wmma_bf16<<<gg, 256, 0, stream>>>(Hb, Wt[3], bs, Ag, N_NODES, dn, Kpad);

    edge_msg<<<dim3(N_EDGES), 256, 0, stream>>>(ei, Kb, Qb, Vb, Ag, dn);

    zero_f32<<<(2 * dn + 255) / 256, 256, 0, stream>>>(st, 2 * dn);
    bn_stats<<<dim3((dn + 255) / 256, (N_NODES + STAT_CHUNK - 1) / STAT_CHUNK),
               256, 0, stream>>>(Ag, st, dn);
    bn_finalize<<<(dn + 255) / 256, 256, 0, stream>>>(st, gamma, beta, dn);

    const int KpadNext = rnd(dn, 32);   // next layer's d_in == this d_out
    bn_apply<<<dim3(N_NODES, (KpadNext + 255) / 256), 256, 0, stream>>>(
        Ag, st, H, Hb, dn, KpadNext);
  }

  zero_f32<<<(out_size + 255) / 256, 256, 0, stream>>>((float*)d_out, out_size);
  pool_sum<<<dim3((1024 + 255) / 256, (N_NODES + POOL_CHUNK - 1) / POOL_CHUNK),
             256, 0, stream>>>(H, batch, (float*)d_out, 1024);
}
